// AlgebraicAttention_17205638988489
// MI455X (gfx1250) — matrix operand order repaired and checked by reading, not compiled, not run on hardware
//
#include <hip/hip_runtime.h>
#include <hip/hip_bf16.h>

// ---------------------------------------------------------------------------
// AlgebraicAttention for MI455X (gfx1250, wave32, WMMA).
//
// Shapes: B=4, T=1024, D=1024, H=16, Dh=64, MAX_REL=32.
// Pipeline (all bf16 matmuls with f32 WMMA accumulate):
//   1) cvt:   x, Wq, Wk, Wv, Wo  -> bf16 in d_ws
//   2) gemm:  Q = xb@Wqb^T + bq   (bf16 out, row-major [B*T][D])
//             K = xb@Wkb^T + bk   (bf16 out, row-major [B*T][D])
//             V = xb@Wvb^T + bv   (bf16 out, TRANSPOSED [(b*H+h)*Dh+d][T])
//   3) attn:  one wave per (b,h,16-query tile); streams keys 32 at a time:
//             s = Q K^T * scale + relbias; w = mask ? relu(s)+1e-6 : 0;
//             rowsum += sum(w);  O += w @ V  (w transposed via LDS);
//             O /= (rowsum + 1e-6)   -> bf16 row-major [B*T][D]
//   4) gemm:  out = Ob@Wob^T + bo  (f32 out -> d_out)
//
// proj_gemm computes a 32x64 tile per wave (8 WMMAs / K-step): every
// B-fragment load is reused by two WMMAs -> 1.5 B128 loads per WMMA.
//
// Workspace layout (48 MB required):
//   [ 0MB) xb  8MB | [ 8MB) Wqb 2MB | [10MB) Wkb 2MB | [12MB) Wvb 2MB
//   [14MB) Wob 2MB | [16MB) Qb  8MB | [24MB) Kb  8MB | [32MB) Vt  8MB
//   [40MB) Ob  8MB
// ---------------------------------------------------------------------------

typedef __attribute__((ext_vector_type(16))) __bf16        v16bf;
typedef __attribute__((ext_vector_type(8)))  float         v8f;
typedef __attribute__((ext_vector_type(8)))  unsigned int  v8u;
typedef __attribute__((ext_vector_type(4)))  unsigned int  v4u;

__device__ __forceinline__ unsigned short f2bf(float f) {
    unsigned int u = __builtin_bit_cast(unsigned int, f);
    u += 0x7fffu + ((u >> 16) & 1u);        // round-to-nearest-even
    return (unsigned short)(u >> 16);
}

__device__ __forceinline__ v8f zero_v8f() {
    v8f z = {0.f, 0.f, 0.f, 0.f, 0.f, 0.f, 0.f, 0.f};
    return z;
}

// Build a 16-bf16 A-fragment from two 16B halves (elems 0..7 | 8..15).
__device__ __forceinline__ v16bf make_frag(v4u a, v4u b) {
    v8u u;
    u[0] = a[0]; u[1] = a[1]; u[2] = a[2]; u[3] = a[3];
    u[4] = b[0]; u[5] = b[1]; u[6] = b[2]; u[7] = b[3];
    return __builtin_bit_cast(v16bf, u);
}

// ---------------------------------------------------------------------------
// f32 -> packed bf16 conversion (2 elements / thread)
// ---------------------------------------------------------------------------
__global__ __launch_bounds__(256) void cvt_f32_bf16(
    const float* __restrict__ src, unsigned int* __restrict__ dst, int npairs) {
    int i = blockIdx.x * 256 + threadIdx.x;
    if (i < npairs) {
        float a = src[2 * i];
        float b = src[2 * i + 1];
        dst[i] = (unsigned int)f2bf(a) | ((unsigned int)f2bf(b) << 16);
    }
}

// ---------------------------------------------------------------------------
// C[m][n] = sum_k A[m][k]*W[n][k] + bias[n]
// One wave computes a 32x64 tile (two 16-row halves share B fragments).
// A: bf16 [M][K], W: bf16 [N][K].
// out_mode 0: bf16 row-major [M][N]
// out_mode 1: bf16 transposed-per-head Vt[((b*16+h)*64+d)*1024 + t]
// out_mode 2: f32 row-major [M][N]
// ---------------------------------------------------------------------------
__global__ __launch_bounds__(32) void proj_gemm(
    const unsigned short* __restrict__ A,
    const unsigned short* __restrict__ W,
    const float* __restrict__ bias,
    void* __restrict__ outp,
    int M, int N, int K, int out_mode) {

    const int lane = threadIdx.x & 31;
    const int ln = lane & 15, hi = lane >> 4;
    const int mt = blockIdx.x, nt = blockIdx.y;
    const size_t arow0 = (size_t)(mt * 32 + ln) * K;
    const size_t arow1 = (size_t)(mt * 32 + 16 + ln) * K;

    v8f acc[2][4];
#pragma unroll
    for (int r = 0; r < 2; ++r)
#pragma unroll
        for (int c = 0; c < 4; ++c) acc[r][c] = zero_v8f();

#pragma unroll 2
    for (int k0 = 0; k0 < K; k0 += 32) {
        // A fragments: lanes<16 hold K k0..k0+15, lanes>=16 hold k0+16..k0+31
        v4u a00 = *(const v4u*)(A + arow0 + k0 + hi * 16);
        v4u a01 = *(const v4u*)(A + arow0 + k0 + hi * 16 + 8);
        v4u a10 = *(const v4u*)(A + arow1 + k0 + hi * 16);
        v4u a11 = *(const v4u*)(A + arow1 + k0 + hi * 16 + 8);
        v16bf af0 = make_frag(a00, a01);
        v16bf af1 = make_frag(a10, a11);
#pragma unroll
        for (int c = 0; c < 4; ++c) {
            int n = nt * 64 + c * 16 + ln;
            // B fragment: lane n holds its column, 16 consecutive K values;
            // reused by both row-halves of the tile.
            v8u bu = *(const v8u*)(W + (size_t)n * K + k0 + hi * 16);
            v16bf bf = __builtin_bit_cast(v16bf, bu);
            acc[0][c] = __builtin_amdgcn_wmma_f32_16x16x32_bf16(
                false, af0, false, bf, (short)0, acc[0][c], false, false);
            acc[1][c] = __builtin_amdgcn_wmma_f32_16x16x32_bf16(
                false, af1, false, bf, (short)0, acc[1][c], false, false);
        }
    }

#pragma unroll
    for (int r = 0; r < 2; ++r) {
#pragma unroll
        for (int c = 0; c < 4; ++c) {
            int col = nt * 64 + c * 16 + ln;
            float bb = bias ? bias[col] : 0.f;
            int mbase = mt * 32 + r * 16 + hi * 8;
            if (out_mode == 0) {
                unsigned short* o = (unsigned short*)outp;
#pragma unroll
                for (int j = 0; j < 8; ++j) {
                    o[(size_t)(mbase + j) * N + col] = f2bf(acc[r][c][j] + bb);
                }
            } else if (out_mode == 1) {
                // V transposed: row = (b*16+h)*64+d, contiguous along t
                unsigned short* o = (unsigned short*)outp;
                int h = col >> 6, d = col & 63;
                int bidx = mbase >> 10;          // T = 1024
                int t0 = mbase & 1023;
                v4u pk;
#pragma unroll
                for (int j = 0; j < 4; ++j) {
                    unsigned int lo = f2bf(acc[r][c][2 * j] + bb);
                    unsigned int hb = f2bf(acc[r][c][2 * j + 1] + bb);
                    pk[j] = lo | (hb << 16);
                }
                *(v4u*)(o + (size_t)((bidx * 16 + h) * 64 + d) * 1024 + t0) = pk;
            } else {
                float* o = (float*)outp;
#pragma unroll
                for (int j = 0; j < 8; ++j) {
                    o[(size_t)(mbase + j) * N + col] = acc[r][c][j] + bb;
                }
            }
        }
    }
}

// ---------------------------------------------------------------------------
// Flash-style masked-relu attention.  grid = (T/16, H, B), block = 32 (1 wave).
// ---------------------------------------------------------------------------
__global__ __launch_bounds__(32) void attn_kernel(
    const unsigned short* __restrict__ Qb,   // bf16 [B*T][D]
    const unsigned short* __restrict__ Kb,   // bf16 [B*T][D]
    const unsigned short* __restrict__ Vt,   // bf16 [(b*16+h)*64+d][T]
    const float* __restrict__ rel_bias,      // f32 [63][16]
    unsigned short* __restrict__ Ob) {       // bf16 [B*T][D]

    constexpr int T = 1024, D = 1024;
    constexpr float scale = 0.125f;          // 64^-0.5

    const int lane = threadIdx.x & 31;
    const int ln = lane & 15, hi = lane >> 4;
    const int qt = blockIdx.x, h = blockIdx.y, b = blockIdx.z;

    __shared__ float sbias[64];              // per-head bias column (63 used)
    __shared__ unsigned short swt[16 * 32];  // w tile for C->A transpose

    if (lane < 63)       sbias[lane]      = rel_bias[lane * 16 + h];
    if (lane + 32 < 63)  sbias[lane + 32] = rel_bias[(lane + 32) * 16 + h];

    // Q fragments for the 16-row query tile (Dh=64 -> two K=32 steps)
    const size_t qrow = (size_t)(b * T + qt * 16 + ln) * D + h * 64;
    v16bf qf[2];
#pragma unroll
    for (int s = 0; s < 2; ++s) {
        v4u a0 = *(const v4u*)(Qb + qrow + s * 32 + hi * 16);
        v4u a1 = *(const v4u*)(Qb + qrow + s * 32 + hi * 16 + 8);
        qf[s] = make_frag(a0, a1);
    }

    v8f oacc[4];
#pragma unroll
    for (int c = 0; c < 4; ++c) oacc[c] = zero_v8f();
    float rsum[8] = {0.f, 0.f, 0.f, 0.f, 0.f, 0.f, 0.f, 0.f};

    for (int ks = 0; ks <= qt * 16; ks += 32) {
        // ---- scores: two 16x16 C tiles over this 32-key chunk ----
        v8f sc[2];
        sc[0] = zero_v8f();
        sc[1] = zero_v8f();
#pragma unroll
        for (int tki = 0; tki < 2; ++tki) {
            int key = ks + tki * 16 + ln;
            int keyc = key < T ? key : T - 1;        // clamp OOB tail (masked later)
            const size_t krow = (size_t)(b * T + keyc) * D + h * 64;
#pragma unroll
            for (int s = 0; s < 2; ++s) {
                v8u bu = *(const v8u*)(Kb + krow + s * 32 + hi * 16);
                v16bf kf = __builtin_bit_cast(v16bf, bu);
                sc[tki] = __builtin_amdgcn_wmma_f32_16x16x32_bf16(
                    false, qf[s], false, kf, (short)0, sc[tki], false, false);
            }
        }

        // ---- bias + causal mask + relu; stage w into LDS (C->A transpose) ----
        float wv[2][8];
#pragma unroll
        for (int tki = 0; tki < 2; ++tki) {
            int key = ks + tki * 16 + ln;
#pragma unroll
            for (int j = 0; j < 8; ++j) {
                int q = qt * 16 + hi * 8 + j;
                int rel = key - q;
                int bkt = rel < -31 ? 0 : (rel > 31 ? 62 : rel + 31);
                float sv = sc[tki][j] * scale + sbias[bkt];
                float w = (key <= q) ? (fmaxf(sv, 0.f) + 1e-6f) : 0.f;
                wv[tki][j] = w;
                swt[(hi * 8 + j) * 32 + tki * 16 + ln] = f2bf(w);
            }
        }

        // ---- running row sums (reduce across the 16-lane C-layout halves) ----
#pragma unroll
        for (int j = 0; j < 8; ++j) {
            float red = wv[0][j] + wv[1][j];
            red += __shfl_xor(red, 1, 32);
            red += __shfl_xor(red, 2, 32);
            red += __shfl_xor(red, 4, 32);
            red += __shfl_xor(red, 8, 32);
            rsum[j] += red;
        }

        asm volatile("s_wait_dscnt 0" ::: "memory");   // LDS w-tile visible

        // ---- O += w @ V ----
        v4u w0 = *(const v4u*)(swt + ln * 32 + hi * 16);
        v4u w1 = *(const v4u*)(swt + ln * 32 + hi * 16 + 8);
        v16bf wf = make_frag(w0, w1);
        int vstart = ks + hi * 16;
        if (vstart > T - 16) vstart = T - 16;          // keep reads in-buffer (w==0 there)
#pragma unroll
        for (int c = 0; c < 4; ++c) {
            const size_t vrow = (size_t)((b * 16 + h) * 64 + c * 16 + ln) * T;
            v8u vu = *(const v8u*)(Vt + vrow + vstart);
            v16bf vf = __builtin_bit_cast(v16bf, vu);
            oacc[c] = __builtin_amdgcn_wmma_f32_16x16x32_bf16(
                false, wf, false, vf, (short)0, oacc[c], false, false);
        }
    }

    // ---- normalize by row sum, store bf16 row-major [B*T][D] ----
#pragma unroll
    for (int c = 0; c < 4; ++c) {
        int col = h * 64 + c * 16 + ln;
#pragma unroll
        for (int j = 0; j < 8; ++j) {
            int m = b * T + qt * 16 + hi * 8 + j;
            float denom = rsum[j] + 1e-6f;
            Ob[(size_t)m * D + col] = f2bf(oacc[c][j] / denom);
        }
    }
}

// ---------------------------------------------------------------------------
extern "C" void kernel_launch(void* const* d_in, const int* in_sizes, int n_in,
                              void* d_out, int out_size, void* d_ws, size_t ws_size,
                              hipStream_t stream) {
    constexpr int B = 4, T = 1024, D = 1024, H = 16;
    constexpr int M = B * T;                       // 4096
    constexpr size_t MB = 1024 * 1024;

    const float* x   = (const float*)d_in[0];
    // d_in[1] = mask (causal, hardcoded)
    const float* Wq  = (const float*)d_in[2];
    const float* bq  = (const float*)d_in[3];
    const float* Wk  = (const float*)d_in[4];
    const float* bk  = (const float*)d_in[5];
    const float* Wv  = (const float*)d_in[6];
    const float* bv  = (const float*)d_in[7];
    const float* Wo  = (const float*)d_in[8];
    const float* bo  = (const float*)d_in[9];
    const float* rel = (const float*)d_in[10];
    // d_in[11] = n_head (hardcoded 16)

    unsigned char* ws = (unsigned char*)d_ws;
    unsigned short* xb  = (unsigned short*)(ws + 0 * MB);
    unsigned short* Wqb = (unsigned short*)(ws + 8 * MB);
    unsigned short* Wkb = (unsigned short*)(ws + 10 * MB);
    unsigned short* Wvb = (unsigned short*)(ws + 12 * MB);
    unsigned short* Wob = (unsigned short*)(ws + 14 * MB);
    unsigned short* Qb  = (unsigned short*)(ws + 16 * MB);
    unsigned short* Kb  = (unsigned short*)(ws + 24 * MB);
    unsigned short* Vt  = (unsigned short*)(ws + 32 * MB);
    unsigned short* Ob  = (unsigned short*)(ws + 40 * MB);

    // 1) bf16 conversions
    {
        int np_x = (M * D) / 2;                     // 2M pairs
        cvt_f32_bf16<<<(np_x + 255) / 256, 256, 0, stream>>>(x, (unsigned int*)xb, np_x);
        int np_w = (D * D) / 2;                     // 512K pairs
        cvt_f32_bf16<<<(np_w + 255) / 256, 256, 0, stream>>>(Wq, (unsigned int*)Wqb, np_w);
        cvt_f32_bf16<<<(np_w + 255) / 256, 256, 0, stream>>>(Wk, (unsigned int*)Wkb, np_w);
        cvt_f32_bf16<<<(np_w + 255) / 256, 256, 0, stream>>>(Wv, (unsigned int*)Wvb, np_w);
        cvt_f32_bf16<<<(np_w + 255) / 256, 256, 0, stream>>>(Wo, (unsigned int*)Wob, np_w);
    }

    // 2) Q/K/V projections (32x64 tiles, one wave each)
    dim3 ggrid(M / 32, D / 64);
    proj_gemm<<<ggrid, 32, 0, stream>>>(xb, Wqb, bq, Qb, M, D, D, 0);
    proj_gemm<<<ggrid, 32, 0, stream>>>(xb, Wkb, bk, Kb, M, D, D, 0);
    proj_gemm<<<ggrid, 32, 0, stream>>>(xb, Wvb, bv, Vt, M, D, D, 1);

    // 3) attention
    attn_kernel<<<dim3(T / 16, H, B), 32, 0, stream>>>(Qb, Kb, Vt, rel, Ob);

    // 4) output projection -> f32 d_out
    proj_gemm<<<ggrid, 32, 0, stream>>>(Ob, Wob, bo, (float*)d_out, M, D, D, 2);
}
